// MoELayer_32950989094983
// MI455X (gfx1250) — compile-verified, matrix operand order checked by reading, not run on hardware
//
#include <hip/hip_runtime.h>
#include <hip/hip_bf16.h>

#define DM 1024   // d_model
#define DF 4096   // d_ff
#define NE 16     // experts
#define NTOK 4096 // B*T
#define FC 512    // f-chunk width

typedef __attribute__((ext_vector_type(16))) __bf16 v16bf;
typedef __attribute__((ext_vector_type(8)))  float  v8f;
typedef __attribute__((ext_vector_type(8)))  unsigned short ushortx8;

union FragBF {
    v16bf v;
    ushortx8 h[2];
};

// ---- fp32 pair -> packed bf16x2 (a -> low half, b -> high half) -------------
#if __has_builtin(__builtin_amdgcn_cvt_pk_bf16_f32)
typedef __attribute__((ext_vector_type(2))) __bf16 bf16x2;
__device__ __forceinline__ unsigned pack2bf(float a, float b) {
    bf16x2 p = __builtin_amdgcn_cvt_pk_bf16_f32(a, b);
    return __builtin_bit_cast(unsigned, p);
}
#else
// round-half-up + v_perm_b32: 3 VALU per pair, no 16-bit sub-reg ops
__device__ __forceinline__ unsigned pack2bf(float a, float b) {
    unsigned ra = __float_as_uint(a) + 0x8000u;
    unsigned rb = __float_as_uint(b) + 0x8000u;
    // dst bytes: [1:0] = ra[3:2], [3:2] = rb[3:2]
    return __builtin_amdgcn_perm(rb, ra, 0x07060302u);
}
#endif

__device__ __forceinline__ unsigned short f2bf(float f) {
    return (unsigned short)((__float_as_uint(f) + 0x8000u) >> 16);
}

// ---------------- Gating: logits -> top-2 -> softmax -> per-expert lists -----
__global__ __launch_bounds__(256) void moe_gate(
    const float* __restrict__ x, const float* __restrict__ Wg,
    int* __restrict__ counts, int* __restrict__ tokList, float* __restrict__ wList)
{
    int wave = threadIdx.x >> 5;
    int lane = threadIdx.x & 31;
    int t = blockIdx.x * 8 + wave;
    if (t >= NTOK) return;

    float acc[NE];
#pragma unroll
    for (int e = 0; e < NE; ++e) acc[e] = 0.f;

    const float* xr = x + (size_t)t * DM;
    for (int d = lane; d < DM; d += 32) {
        float xv = xr[d];
        const float4* wg = (const float4*)(Wg + (size_t)d * NE);
#pragma unroll
        for (int q = 0; q < 4; ++q) {
            float4 w = wg[q];
            acc[q * 4 + 0] += xv * w.x;
            acc[q * 4 + 1] += xv * w.y;
            acc[q * 4 + 2] += xv * w.z;
            acc[q * 4 + 3] += xv * w.w;
        }
    }
#pragma unroll
    for (int off = 16; off >= 1; off >>= 1) {
#pragma unroll
        for (int e = 0; e < NE; ++e)
            acc[e] += __shfl_down(acc[e], off, 32);
    }

    if (lane == 0) {
        int i0 = 0; float v0 = acc[0];
#pragma unroll
        for (int e = 1; e < NE; ++e) if (acc[e] > v0) { v0 = acc[e]; i0 = e; }
        int i1 = (i0 == 0) ? 1 : 0; float v1 = acc[i1];
#pragma unroll
        for (int e = 0; e < NE; ++e)
            if (e != i0 && acc[e] > v1) { v1 = acc[e]; i1 = e; }
        float e1 = __expf(v1 - v0);      // v0 is the max
        float s  = 1.f + e1;
        float w0 = 1.f / s, w1 = e1 / s;
        int p0 = atomicAdd(&counts[i0], 1);
        tokList[i0 * NTOK + p0] = t;  wList[i0 * NTOK + p0] = w0;
        int p1 = atomicAdd(&counts[i1], 1);
        tokList[i1 * NTOK + p1] = t;  wList[i1 * NTOK + p1] = w1;
    }
}

// ---------------- Routed expert FFN: out += w * (swish(xW1+b1) W2 + b2) ------
//
// Weight chunks staged through LDS in WMMA-B-fragment-ready layout:
//   stage[ntile][lane][j]  (16 bf16 per lane = 32 B, read as 2x ds_load_b128)
// K <-> (lane-half, j): hi = (K>>3)&1, j = (K&7) + 8*(K>>4).
// Even-k pairs (k,k+1) share a lane at adjacent j -> packed b32 stores.
__global__ __launch_bounds__(256, 1) void moe_ffn(
    const float* __restrict__ x,  const float* __restrict__ W1,
    const float* __restrict__ b1, const float* __restrict__ W2,
    const float* __restrict__ b2,
    const int*   __restrict__ counts, const int* __restrict__ tokList,
    const float* __restrict__ wList,  float* __restrict__ out)
{
    __shared__ unsigned short x_lds[16][DM];   // 32 KB  bf16 A-operand (GEMM1)
    __shared__ unsigned short h_lds[16][FC];   // 16 KB  bf16 A-operand (GEMM2)
    __shared__ unsigned short stage[64 * 512]; // 64 KB  swizzled B staging
    __shared__ int   tok_s[16];
    __shared__ float w_s[16];

    int e   = blockIdx.y;
    int cnt = counts[e];
    int m0  = blockIdx.x * 16;
    if (m0 >= cnt) return;

    int tid  = threadIdx.x;
    int wave = tid >> 5;
    int lane = tid & 31;

    if (tid < 16) {
        int idx = m0 + tid;
        if (idx < cnt) {
            tok_s[tid] = tokList[e * NTOK + idx];
            w_s[tid]   = wList[e * NTOK + idx];
        } else {
            tok_s[tid] = 0;
            w_s[tid]   = 0.f;    // padded row: contributes nothing
        }
    }
    __syncthreads();

    // Gather 16 token rows of x into LDS as bf16 (float4 reads, packed stores)
    {
        unsigned* xl32 = (unsigned*)&x_lds[0][0];
#pragma unroll
        for (int it = 0; it < 16; ++it) {
            int vid = it * 256 + tid;        // 4096 float4s total
            int r  = vid >> 8;               // token row
            int d4 = vid & 255;              // float4 index along d
            float4 fv = *(const float4*)(x + (size_t)tok_s[r] * DM + d4 * 4);
            xl32[(r << 9) + d4 * 2 + 0] = pack2bf(fv.x, fv.y);
            xl32[(r << 9) + d4 * 2 + 1] = pack2bf(fv.z, fv.w);
        }
    }

    const float* W1e = W1 + (size_t)e * DM * DF;
    const float* W2e = W2 + (size_t)e * DF * DM;
    unsigned* s32 = (unsigned*)stage;

    v8f acc2[8];
#pragma unroll
    for (int t = 0; t < 8; ++t) acc2[t] = (v8f){0.f,0.f,0.f,0.f,0.f,0.f,0.f,0.f};

    const int mrow  = lane & 15;            // A row index
    const int khalf = (lane < 16) ? 0 : 8;  // per-half-wave K base (A layout)
    const int mbase = (lane < 16) ? 0 : 8;  // C/D row base

    for (int fc = 0; fc < DF; fc += FC) {
        // ---- GEMM1: h_chunk = swish(x_tile @ W1[:, fc:fc+FC] + b1) ----------
        v8f acc1[4];
#pragma unroll
        for (int t = 0; t < 4; ++t) acc1[t] = (v8f){0.f,0.f,0.f,0.f,0.f,0.f,0.f,0.f};

        for (int k0 = 0; k0 < DM; k0 += 32) {
            __syncthreads();   // staging buffer free (prev k-step consumed)
            // stage 32(K) x 512(N): 2048 octs (2 float4 rows -> 4 packed b32)
#pragma unroll
            for (int it = 0; it < 8; ++it) {
                int oid = it * 256 + tid;
                int n4  = oid & 127;            // float4 index along N
                int kp  = oid >> 7;             // 0..15
                int k   = kp * 2;
                const float* bp0 = W1e + (size_t)(k0 + k) * DF + fc + n4 * 4;
                float4 a = *(const float4*)bp0;        // row k
                float4 b = *(const float4*)(bp0 + DF); // row k+1
                int hi = (k >> 3) & 1;
                int jp = ((k & 7) + ((k >> 4) << 3)) >> 1;   // 0..7
                int n  = n4 * 4;                 // n..n+3 share ntile (n%16 in {0,4,8,12})
                int t0 = ((n >> 4) << 8) + (((n & 15) + (hi << 4)) << 3) + jp;
                s32[t0     ] = pack2bf(a.x, b.x);
                s32[t0 +  8] = pack2bf(a.y, b.y);
                s32[t0 + 16] = pack2bf(a.z, b.z);
                s32[t0 + 24] = pack2bf(a.w, b.w);
            }
            __syncthreads();   // staged data visible (also covers x_lds on k0==0)

            FragBF A;
            A.h[0] = *(const ushortx8*)&x_lds[mrow][k0 + khalf];
            A.h[1] = *(const ushortx8*)&x_lds[mrow][k0 + khalf + 16];
#pragma unroll
            for (int t = 0; t < 4; ++t) {
                int ntile = wave * 4 + t;
                const ushortx8* bp = (const ushortx8*)&stage[ntile * 512 + lane * 16];
                FragBF Bf;
                Bf.h[0] = bp[0];
                Bf.h[1] = bp[1];
                acc1[t] = __builtin_amdgcn_wmma_f32_16x16x32_bf16(
                    false, A.v, false, Bf.v, (short)0, acc1[t], false, false);
            }
        }
        // epilogue: + b1, swish (fast rcp), pack to bf16 h tile in LDS
        __syncthreads();
#pragma unroll
        for (int t = 0; t < 4; ++t) {
            int n  = wave * 64 + t * 16 + mrow;
            float bb = b1[e * DF + fc + n];
#pragma unroll
            for (int r = 0; r < 8; ++r) {
                float v  = acc1[t][r] + bb;
                float sw = v * __builtin_amdgcn_rcpf(1.f + __expf(-v));  // swish
                h_lds[mbase + r][n] = f2bf(sw);
            }
        }

        // ---- GEMM2: acc2 += h_chunk @ W2[fc:fc+FC, :] -----------------------
        for (int k0 = 0; k0 < FC; k0 += 32) {
            __syncthreads();   // staging free; first iter also covers h_lds writes
            // stage 32(K) x 1024(N): 4096 octs
#pragma unroll
            for (int it = 0; it < 16; ++it) {
                int oid = it * 256 + tid;
                int n4  = oid & 255;
                int kp  = oid >> 8;             // 0..15
                int k   = kp * 2;
                const float* bp0 = W2e + (size_t)(fc + k0 + k) * DM + n4 * 4;
                float4 a = *(const float4*)bp0;        // row k
                float4 b = *(const float4*)(bp0 + DM); // row k+1
                int hi = (k >> 3) & 1;
                int jp = ((k & 7) + ((k >> 4) << 3)) >> 1;
                int n  = n4 * 4;
                int t0 = ((n >> 4) << 8) + (((n & 15) + (hi << 4)) << 3) + jp;
                s32[t0     ] = pack2bf(a.x, b.x);
                s32[t0 +  8] = pack2bf(a.y, b.y);
                s32[t0 + 16] = pack2bf(a.z, b.z);
                s32[t0 + 24] = pack2bf(a.w, b.w);
            }
            __syncthreads();

            FragBF A;
            A.h[0] = *(const ushortx8*)&h_lds[mrow][k0 + khalf];
            A.h[1] = *(const ushortx8*)&h_lds[mrow][k0 + khalf + 16];
#pragma unroll
            for (int t = 0; t < 8; ++t) {
                int ntile = wave * 8 + t;
                const ushortx8* bp = (const ushortx8*)&stage[ntile * 512 + lane * 16];
                FragBF Bf;
                Bf.h[0] = bp[0];
                Bf.h[1] = bp[1];
                acc2[t] = __builtin_amdgcn_wmma_f32_16x16x32_bf16(
                    false, A.v, false, Bf.v, (short)0, acc2[t], false, false);
            }
        }
        __syncthreads();   // h_lds rewritten next chunk
    }

    // ---- combine: + b2, scale by gate weight, scatter-add into output ------
#pragma unroll
    for (int t = 0; t < 8; ++t) {
        int n  = wave * 128 + t * 16 + mrow;
        float bb = b2[e * DM + n];
#pragma unroll
        for (int r = 0; r < 8; ++r) {
            int m = mbase + r;
            float w = w_s[m];
            if (w != 0.f)
                atomicAdd(&out[(size_t)tok_s[m] * DM + n], w * (acc2[t][r] + bb));
        }
    }
}

extern "C" void kernel_launch(void* const* d_in, const int* in_sizes, int n_in,
                              void* d_out, int out_size, void* d_ws, size_t ws_size,
                              hipStream_t stream) {
    const float* x  = (const float*)d_in[0];
    const float* W1 = (const float*)d_in[1];
    const float* b1 = (const float*)d_in[2];
    const float* W2 = (const float*)d_in[3];
    const float* b2 = (const float*)d_in[4];
    const float* Wg = (const float*)d_in[5];
    // d_in[6] = k (==2, hardcoded)

    char* ws = (char*)d_ws;
    int*   counts  = (int*)ws;                                     // 16 ints
    int*   tokList = (int*)(ws + 256);                             // 16*4096 ints
    float* wList   = (float*)(ws + 256 + NE * NTOK * sizeof(int)); // 16*4096 floats
    float* out     = (float*)d_out;

    hipMemsetAsync(counts, 0, 64, stream);
    hipMemsetAsync(out, 0, (size_t)out_size * sizeof(float), stream);

    moe_gate<<<NTOK / 8, 256, 0, stream>>>(x, Wg, counts, tokList, wList);
    moe_ffn<<<dim3(256, NE, 1), 256, 0, stream>>>(
        x, W1, b1, W2, b2, counts, tokList, wList, out);
}
